// TextualEmbeddingLayer_24026047054349
// MI455X (gfx1250) — matrix operand order repaired and checked by reading, not compiled, not run on hardware
//
#include <hip/hip_runtime.h>
#include <math.h>

typedef unsigned short u16;
typedef __attribute__((ext_vector_type(16))) __bf16 bf16x16;
typedef __attribute__((ext_vector_type(8)))  float  f32x8;

#define BB 32
#define LL 512
#define DD 512
#define EE 4096
#define HH 2048
#define KK 204
#define MROWS (BB * KK)   // 6528

union Frag {
  uint4   q[2];
  bf16x16 v;
};

__device__ inline u16 f2bf(float f) {
  unsigned u = __float_as_uint(f);
  unsigned r = u + 0x7FFFu + ((u >> 16) & 1u);   // round-to-nearest-even
  return (u16)(r >> 16);
}
__device__ inline float bf2f(u16 h) {
  return __uint_as_float(((unsigned)h) << 16);
}

__device__ inline void atomicMaxF32(float* addr, float val) {
  unsigned* ua = (unsigned*)addr;
  unsigned cur = *(volatile unsigned*)ua;
  while (__uint_as_float(cur) < val) {
    unsigned prev = atomicCAS(ua, cur, __float_as_uint(val));
    if (prev == cur) break;
    cur = prev;
  }
}

// ---------------------------------------------------------------------------
// 1) Per-batch prep: mask, lengths, EOS argmax, knockout, full bitonic sort
//    (descending score, tie -> lower index, matching jax.lax.top_k)
// ---------------------------------------------------------------------------
__global__ __launch_bounds__(512)
void prep_topk(const int* __restrict__ text, const float* __restrict__ attention,
               int* __restrict__ idxbuf, int* __restrict__ lengths) {
  const int b = blockIdx.x;
  const int tid = threadIdx.x;
  __shared__ float sv[LL];
  __shared__ int   si[LL];
  __shared__ int   scnt[LL];

  const int t = text[b * LL + tid];
  const int m = (t != 0) ? 1 : 0;
  scnt[tid] = m;
  sv[tid] = (float)t;      // 0..49407 exact in f32
  si[tid] = tid;
  __syncthreads();
  for (int s = 256; s > 0; s >>= 1) {
    if (tid < s) {
      scnt[tid] += scnt[tid + s];
      if (sv[tid + s] > sv[tid] ||
          (sv[tid + s] == sv[tid] && si[tid + s] < si[tid])) {
        sv[tid] = sv[tid + s];
        si[tid] = si[tid + s];
      }
    }
    __syncthreads();
  }
  const int eos = si[0];
  const int total = scnt[0];
  __syncthreads();

  float r = attention[((size_t)b * LL + eos) * LL + tid];
  if (tid == eos || tid == 0) r = -1.0f;
  r *= (float)m;
  sv[tid] = r;
  si[tid] = tid;
  __syncthreads();

  // bitonic sort: final order = descending value, ascending index on ties
  for (int k = 2; k <= LL; k <<= 1) {
    for (int j = k >> 1; j > 0; j >>= 1) {
      const int ixj = tid ^ j;
      if (ixj > tid) {
        const bool dirAsc = ((tid & k) == 0);
        const float va = sv[tid], vb = sv[ixj];
        const int ia = si[tid], ib = si[ixj];
        const bool after = (va < vb) || (va == vb && ia > ib);
        if (after == dirAsc) {
          sv[tid] = vb; si[tid] = ib;
          sv[ixj] = va; si[ixj] = ia;
        }
      }
      __syncthreads();
    }
  }
  if (tid < KK) idxbuf[b * KK + tid] = si[tid];
  if (tid == 0) {
    int len = total - 2;
    if (len < 1) len = 1;
    if (len > KK) len = KK;
    lengths[b] = len;
  }
}

// ---------------------------------------------------------------------------
// 2) Gather selected tokens, L2-normalize, convert to bf16
// ---------------------------------------------------------------------------
__global__ __launch_bounds__(256)
void gather_norm(const float* __restrict__ features, const int* __restrict__ idxbuf,
                 u16* __restrict__ selA) {
  const int mrow = blockIdx.x;            // 0..6527
  const int b = mrow / KK;
  const int src = idxbuf[mrow];
  const float* fp = features + ((size_t)b * LL + src) * DD;
  const float v0 = fp[threadIdx.x];
  const float v1 = fp[threadIdx.x + 256];
  __shared__ float red[256];
  red[threadIdx.x] = v0 * v0 + v1 * v1;
  __syncthreads();
  for (int s = 128; s > 0; s >>= 1) {
    if (threadIdx.x < s) red[threadIdx.x] += red[threadIdx.x + s];
    __syncthreads();
  }
  const float nrm = 1.0f / (sqrtf(red[0]) + 1e-8f);
  u16* op = selA + (size_t)mrow * DD;
  op[threadIdx.x]       = f2bf(v0 * nrm);
  op[threadIdx.x + 256] = f2bf(v1 * nrm);
}

// ---------------------------------------------------------------------------
// 3) f32 -> bf16 conversion (weights)
// ---------------------------------------------------------------------------
__global__ __launch_bounds__(256)
void cvt_bf16(const float* __restrict__ src, u16* __restrict__ dst, int n) {
  for (size_t i = (size_t)blockIdx.x * blockDim.x + threadIdx.x; i < (size_t)n;
       i += (size_t)gridDim.x * blockDim.x)
    dst[i] = f2bf(src[i]);
}

// ---------------------------------------------------------------------------
// WMMA core: per-wave 32x64 tile, bf16 x bf16 -> f32, D = A*W^T.
// A layout (16-bit 16x32): lane<16 holds row m=lane, halves K[kb..kb+7],K[kb+16..kb+23]
//                          with kb = 8*(lane>=16).
// B layout (16-bit 32x16): lane holds column n=lane%16, halves K[kb..kb+7],K[kb+16..kb+23]
//                          with kb = k0 + 8*(lane>=16). Column n of B is row n
//                          of the (row-major, K-contiguous) weight matrix.
// ---------------------------------------------------------------------------
__device__ inline void wmma_loop32(const u16* __restrict__ A, const u16* __restrict__ W,
                                   int K, int tm, int tn, int lane, f32x8 acc[2][4]) {
  const int lane15 = lane & 15;
  const int hi = (lane >> 4) & 1;
  const u16* aBase0 = A + (size_t)(tm + lane15) * K + hi * 8;
  const u16* aBase1 = aBase0 + (size_t)16 * K;
  const int boff = hi * 8;
  for (int k0 = 0; k0 < K; k0 += 32) {
    Frag fa0, fa1;
    fa0.q[0] = *(const uint4*)(aBase0 + k0);
    fa0.q[1] = *(const uint4*)(aBase0 + k0 + 16);
    fa1.q[0] = *(const uint4*)(aBase1 + k0);
    fa1.q[1] = *(const uint4*)(aBase1 + k0 + 16);
#pragma unroll
    for (int a = 0; a < 4; ++a) {
      const u16* bp = W + (size_t)(tn + a * 16 + lane15) * K + boff + k0;
      Frag fb;
      fb.q[0] = *(const uint4*)bp;
      fb.q[1] = *(const uint4*)(bp + 16);
      acc[0][a] = __builtin_amdgcn_wmma_f32_16x16x32_bf16(
          false, fa0.v, false, fb.v, (short)0, acc[0][a], false, false);
      acc[1][a] = __builtin_amdgcn_wmma_f32_16x16x32_bf16(
          false, fa1.v, false, fb.v, (short)0, acc[1][a], false, false);
    }
  }
}

// Block tiling: 8 waves arranged 4 (M) x 2 (N); block covers 128 rows x 128 cols.
// All 8 waves touch the same A rows / B columns region -> WGP$ reuse, ~2x less
// L2 traffic vs a 1D wave mapping.
__device__ inline void block_tile(int tilesN64, int* tm, int* tn) {
  const int NB = tilesN64 >> 1;              // block-columns
  const int bm = blockIdx.x / NB;
  const int bn = blockIdx.x % NB;
  const int w = threadIdx.x >> 5;            // 0..7
  *tm = (bm * 4 + (w >> 1)) * 32;
  *tn = (bn * 2 + (w & 1)) * 64;
}

// ---------------------------------------------------------------------------
// 4) hidden = sel @ W1^T + b1  -> bf16 [6528, 2048]
// ---------------------------------------------------------------------------
__global__ __launch_bounds__(256)
void gemm_hidden(const u16* __restrict__ selA, const u16* __restrict__ w1b,
                 const float* __restrict__ b1, u16* __restrict__ hidden) {
  const int N = HH, K = DD;
  int tm, tn;
  block_tile(N / 64, &tm, &tn);
  const int lane = threadIdx.x & 31;

  f32x8 acc[2][4] = {};
  wmma_loop32(selA, w1b, K, tm, tn, lane, acc);

  const int lane15 = lane & 15;
  const int hi = (lane >> 4) & 1;
#pragma unroll
  for (int rg = 0; rg < 2; ++rg) {
#pragma unroll
    for (int a = 0; a < 4; ++a) {
      const int n = tn + a * 16 + lane15;
      const float bias = b1[n];
#pragma unroll
      for (int r = 0; r < 8; ++r) {
        const int mm = tm + rg * 16 + hi * 8 + r;     // D layout: VGPR r -> row r + 8*hi
        hidden[(size_t)mm * N + n] = f2bf(acc[rg][a][r] + bias);
      }
    }
  }
}

// ---------------------------------------------------------------------------
// 5) BatchNorm batch stats -> per-column scale/shift
// ---------------------------------------------------------------------------
__global__ __launch_bounds__(256)
void bn_stats(const u16* __restrict__ hidden, const float* __restrict__ g1,
              const float* __restrict__ beta1, float* __restrict__ scale,
              float* __restrict__ shift) {
  const int col = blockIdx.x * blockDim.x + threadIdx.x;
  if (col >= HH) return;
  float s = 0.f, ss = 0.f;
  const u16* p = hidden + col;
  for (int i = 0; i < MROWS; ++i) {
    const float v = bf2f(p[(size_t)i * HH]);
    s += v;
    ss += v * v;
  }
  const float mean = s / (float)MROWS;
  const float var = ss / (float)MROWS - mean * mean;   // biased, matches torch BN
  const float inv = rsqrtf(var + 1e-5f);
  const float sc = g1[col] * inv;
  scale[col] = sc;
  shift[col] = beta1[col] - mean * sc;
}

// 6) y = relu(x*scale + shift), in place (bf16)
__global__ __launch_bounds__(256)
void bn_relu(u16* __restrict__ hidden, const float* __restrict__ scale,
             const float* __restrict__ shift) {
  const size_t total = (size_t)MROWS * HH;
  for (size_t i = (size_t)blockIdx.x * blockDim.x + threadIdx.x; i < total;
       i += (size_t)gridDim.x * blockDim.x) {
    const int col = (int)(i & (HH - 1));
    const float x = bf2f(hidden[i]);
    hidden[i] = f2bf(fmaxf(x * scale[col] + shift[col], 0.0f));
  }
}

// 7) out = -inf
__global__ __launch_bounds__(256)
void init_out(float* __restrict__ out) {
  out[(size_t)blockIdx.x * blockDim.x + threadIdx.x] = -INFINITY;
}

// ---------------------------------------------------------------------------
// 8) refined = sel@W_lin^T + y@W2^T + (b_lin+b2); fused masked max-pool into
//    d_out. Rows covered by one lane span at most 2 batches, so max-reduce in
//    registers and flush <=2 atomics per (lane, column).
// ---------------------------------------------------------------------------
__global__ __launch_bounds__(256)
void gemm_fused_pool(const u16* __restrict__ selA, const u16* __restrict__ wlinb,
                     const u16* __restrict__ yb, const u16* __restrict__ w2b,
                     const float* __restrict__ b_lin, const float* __restrict__ b2,
                     const int* __restrict__ lengths, float* __restrict__ out) {
  int tm, tn;
  block_tile(EE / 64, &tm, &tn);
  const int lane = threadIdx.x & 31;

  f32x8 acc[2][4] = {};
  wmma_loop32(selA, wlinb, DD, tm, tn, lane, acc);   // base projection, K=512
  wmma_loop32(yb, w2b, HH, tm, tn, lane, acc);       // MLP second layer, K=2048

  const int lane15 = lane & 15;
  const int hi = (lane >> 4) & 1;
#pragma unroll
  for (int a = 0; a < 4; ++a) {
    const int n = tn + a * 16 + lane15;
    const float bias = b_lin[n] + b2[n];
    float best = -INFINITY;
    int curb = -1;
#pragma unroll
    for (int rg = 0; rg < 2; ++rg) {
#pragma unroll
      for (int r = 0; r < 8; ++r) {
        const int mm = tm + rg * 16 + hi * 8 + r;   // ascending across (rg, r)
        const int bb = mm / KK;
        const int jj = mm - bb * KK;
        if (bb != curb) {
          if (curb >= 0 && best > -INFINITY)
            atomicMaxF32(out + (size_t)curb * EE + n, best);
          curb = bb;
          best = -INFINITY;
        }
        if (jj < lengths[bb]) best = fmaxf(best, acc[rg][a][r] + bias);
      }
    }
    if (curb >= 0 && best > -INFINITY)
      atomicMaxF32(out + (size_t)curb * EE + n, best);
  }
}

// ---------------------------------------------------------------------------
extern "C" void kernel_launch(void* const* d_in, const int* in_sizes, int n_in,
                              void* d_out, int out_size, void* d_ws, size_t ws_size,
                              hipStream_t stream) {
  (void)in_sizes; (void)n_in; (void)out_size; (void)ws_size;
  const float* features  = (const float*)d_in[0];
  const int*   text      = (const int*)d_in[1];
  const float* attention = (const float*)d_in[2];
  const float* W_lin     = (const float*)d_in[3];
  const float* b_lin     = (const float*)d_in[4];
  const float* W1        = (const float*)d_in[5];
  const float* b1        = (const float*)d_in[6];
  const float* g1        = (const float*)d_in[7];
  const float* beta1     = (const float*)d_in[8];
  const float* W2        = (const float*)d_in[9];
  const float* b2        = (const float*)d_in[10];
  float* out = (float*)d_out;

  char* ws = (char*)d_ws;
  size_t off = 0;
  auto take = [&](size_t bytes) -> char* {
    char* p = ws + off;
    off = (off + bytes + 255) & ~(size_t)255;
    return p;
  };
  int* idxbuf  = (int*)take((size_t)MROWS * sizeof(int));
  int* lengths = (int*)take(BB * sizeof(int));
  u16* selA    = (u16*)take((size_t)MROWS * DD * 2);   // 6.7 MB
  u16* wlinb   = (u16*)take((size_t)EE * DD * 2);      // 4 MB
  u16* w1b     = (u16*)take((size_t)HH * DD * 2);      // 2 MB
  u16* w2b     = (u16*)take((size_t)EE * HH * 2);      // 16 MB
  u16* hidden  = (u16*)take((size_t)MROWS * HH * 2);   // 26.7 MB
  float* scale = (float*)take(HH * sizeof(float));
  float* shift = (float*)take(HH * sizeof(float));

  prep_topk<<<BB, LL, 0, stream>>>(text, attention, idxbuf, lengths);
  gather_norm<<<MROWS, 256, 0, stream>>>(features, idxbuf, selA);
  cvt_bf16<<<2048, 256, 0, stream>>>(W_lin, wlinb, EE * DD);
  cvt_bf16<<<1024, 256, 0, stream>>>(W1, w1b, HH * DD);
  cvt_bf16<<<4096, 256, 0, stream>>>(W2, w2b, EE * HH);
  // blocks: (204/4 M-groups) x (tilesN/2 N-groups)
  gemm_hidden<<<(MROWS / 128) * (HH / 128), 256, 0, stream>>>(selA, w1b, b1, hidden);
  bn_stats<<<HH / 256, 256, 0, stream>>>(hidden, g1, beta1, scale, shift);
  bn_relu<<<4096, 256, 0, stream>>>(hidden, scale, shift);
  init_out<<<(BB * EE) / 256, 256, 0, stream>>>(out);
  gemm_fused_pool<<<(MROWS / 128) * (EE / 128), 256, 0, stream>>>(
      selA, wlinb, hidden, w2b, b_lin, b2, lengths, out);
}
